// LSTM_54984171323632
// MI455X (gfx1250) — matrix operand order repaired and checked by reading, not compiled, not run on hardware
//
#include <hip/hip_runtime.h>

typedef __attribute__((ext_vector_type(16))) _Float16 v16h;
typedef __attribute__((ext_vector_type(8)))  float    v8f;

#define DEV __device__ __forceinline__

namespace {

constexpr int T_STEPS = 512;
constexpr int IN_DIM  = 50;
constexpr int HID     = 32;

union F16Frag { v16h v; _Float16 h[16]; uint4 q[2]; };
union F32Acc  { v8f  v; float f[8]; };

DEV float rcp_f(float x) { return __builtin_amdgcn_rcpf(x); }
DEV float ex2_f(float x) { return __builtin_amdgcn_exp2f(x); }

#if __has_builtin(__builtin_amdgcn_tanhf)
#define HAVE_NATIVE_TANH 1
DEV float tanh_fast(float x) { return __builtin_amdgcn_tanhf(x); }
#elif __has_builtin(__builtin_amdgcn_tanh_f32)
#define HAVE_NATIVE_TANH 1
DEV float tanh_fast(float x) { return __builtin_amdgcn_tanh_f32(x); }
#else
#define HAVE_NATIVE_TANH 0
DEV float tanh_fast(float x) { return 1.0f - 2.0f * rcp_f(1.0f + ex2_f(2.88539008f * x)); }
#endif

// sigmoid given the PRE-HALVED argument z = x/2 (0.5 folded into weights/bias):
// sigmoid(x) = 0.5*tanh(x/2) + 0.5
#if HAVE_NATIVE_TANH
DEV float sigm_pre(float z) { return __builtin_fmaf(0.5f, tanh_fast(z), 0.5f); }
#else
DEV float sigm_pre(float z) { return rcp_f(1.0f + ex2_f(-2.88539008f * z)); }
#endif

// B-matrix fragment (32x16 f16) of scale * W^T, where W is [128][kdim] row-major.
// B lane layout: lane L holds col n=L&15; K = 8*(L>>4)+{0..7} in h[0..7], 16+8*(L>>4)+{0..7} in h[8..15].
DEV v16h load_bfrag(const float* W, int kdim, int ntile, int kc, int lane, float scale) {
  const int n     = ntile * 16 + (lane & 15);
  const int kbase = kc * 32 + (lane >> 4) * 8;
  const float* row = W + (size_t)n * kdim;
  F16Frag r;
#pragma unroll
  for (int i = 0; i < 16; ++i) {
    const int k = kbase + (i < 8 ? i : 8 + i);
    r.h[i] = (_Float16)((k < kdim) ? scale * row[k] : 0.0f);
  }
  return r.v;
}

DEV void load8(const float* p, _Float16* dst) {
#pragma unroll
  for (int i = 0; i < 4; ++i) {
    float2 v = *(const float2*)(p + 2 * i);   // x rows are only 8B aligned (stride 200B)
    dst[2 * i]     = (_Float16)v.x;
    dst[2 * i + 1] = (_Float16)v.y;
  }
}

// A-matrix fragments (16x32 f16 each) for one timestep of the x tile.
// A lane layout: lane L holds row M=L&15; K = 8*(L>>4)+{0..7} in h[0..7], 16+8*(L>>4)+{0..7} in h[8..15].
DEV void load_xfrags(const float* xt, int lane, v16h& a0, v16h& a1) {
  const int m  = lane & 15;
  const int ho = (lane >> 4) * 8;
  const float* row = xt + (size_t)m * (T_STEPS * IN_DIM);
  F16Frag f0, f1;
  load8(row + ho,      &f0.h[0]);   // K = ho..ho+7       (<=15)
  load8(row + 16 + ho, &f0.h[8]);   // K = 16+ho..        (<=31)
  load8(row + 32 + ho, &f1.h[0]);   // K = 32+ho..        (<=47)
#pragma unroll
  for (int i = 0; i < 8; ++i) f1.h[8 + i] = (_Float16)0.0f;
  if (ho == 0) {                    // K = 48,49 valid only for lane-half 0
    float2 tt = *(const float2*)(row + 48);
    f1.h[8] = (_Float16)tt.x;
    f1.h[9] = (_Float16)tt.y;
  }
  a0 = f0.v;
  a1 = f1.v;
}

DEV v8f wmma_f16(v16h a, v16h b, v8f c) {
  return __builtin_amdgcn_wmma_f32_16x16x32_f16(false, a, false, b, (short)0, c, false, false);
}

// LDS fragment-cache slot helpers (each slot: 32 lanes x 16 halfs = 512 halfs = 1 KB)
DEV void stash_frag(_Float16* base, int slot, int lane, v16h v) {
  F16Frag f; f.v = v;
  *(uint4*)&base[slot * 512 + lane * 16]     = f.q[0];
  *(uint4*)&base[slot * 512 + lane * 16 + 8] = f.q[1];
}
DEV v16h fetch_frag(const _Float16* base, int slot, int lane) {
  F16Frag f;
  f.q[0] = *(const uint4*)&base[slot * 512 + lane * 16];
  f.q[1] = *(const uint4*)&base[slot * 512 + lane * 16 + 8];
  return f.v;
}

__global__ __launch_bounds__(64)
void lstm2_fused(const float* __restrict__ x,
                 const float* __restrict__ Wih0, const float* __restrict__ Whh0,
                 const float* __restrict__ bih0, const float* __restrict__ bhh0,
                 const float* __restrict__ Wih1, const float* __restrict__ Whh1,
                 const float* __restrict__ bih1, const float* __restrict__ bhh1,
                 const float* __restrict__ fc_w, const float* __restrict__ fc_b,
                 float* __restrict__ out) {
  // Slots 0..15 : Wih0 (ntile*2 + kchunk), 16..23 : Wih1 (16+ntile), 24..31 : Whh1 (24+ntile)
  __shared__ __align__(16) _Float16 sh_wcache[32 * 512];   // 32 KB
  __shared__ __align__(16) _Float16 sh_h0[16 * 32];
  __shared__ __align__(16) _Float16 sh_h1[16 * 32];
  __shared__ __align__(16) float    sh_f [16 * 32];

  const int lane = threadIdx.x & 31;
  const int wv   = threadIdx.x >> 5;   // wave 0/1 owns hidden cols [16*wv, 16*wv+16)
  const int b0   = blockIdx.x * 16;    // batch tile

  // ---- weights: Whh0 fragments resident in VGPRs; the rest staged to LDS.
  //      Sigmoid gates (i,f,o) are pre-scaled by 0.5 (sigm_pre consumes halved args). ----
  v16h whh0B[4];
  float bias0[4], bias1[4];
#pragma unroll
  for (int G = 0; G < 4; ++G) {        // G = gate (i,f,g,o); N-tile = 2G + wv
    const int nt = 2 * G + wv;
    const float sc = (G == 2) ? 1.0f : 0.5f;
    stash_frag(sh_wcache, nt * 2 + 0, lane, load_bfrag(Wih0, IN_DIM, nt, 0, lane, sc));
    stash_frag(sh_wcache, nt * 2 + 1, lane, load_bfrag(Wih0, IN_DIM, nt, 1, lane, sc));
    stash_frag(sh_wcache, 16 + nt,    lane, load_bfrag(Wih1, HID,    nt, 0, lane, sc));
    stash_frag(sh_wcache, 24 + nt,    lane, load_bfrag(Whh1, HID,    nt, 0, lane, sc));
    whh0B[G] = load_bfrag(Whh0, HID, nt, 0, lane, sc);
    const int n = nt * 16 + (lane & 15);
    bias0[G] = sc * (bih0[n] + bhh0[n]);
    bias1[G] = sc * (bih1[n] + bhh1[n]);
  }

  // ---- recurrent state ----
  float c0[8], c1[8], h1last[8];
#pragma unroll
  for (int r = 0; r < 8; ++r) { c0[r] = 0.f; c1[r] = 0.f; h1last[r] = 0.f; }
  F16Frag h0f, h1f;
#pragma unroll
  for (int i = 0; i < 16; ++i) { h0f.h[i] = (_Float16)0.f; h1f.h[i] = (_Float16)0.f; }

  const int mrow = lane & 15;              // A-frag row this lane reads
  const int half = lane >> 4;              // C-layout row half this lane holds
  const int hcol = 16 * wv + (lane & 15);  // hidden column this lane owns

  const float* xbase = x + (size_t)b0 * T_STEPS * IN_DIM;
  v16h xa0, xa1, nx0, nx1;
  load_xfrags(xbase, lane, xa0, xa1);

#pragma unroll 1
  for (int t = 0; t < T_STEPS; ++t) {
    if (t + 1 < T_STEPS)                   // software prefetch next timestep
      load_xfrags(xbase + (size_t)(t + 1) * IN_DIM, lane, nx0, nx1);

    // ---- layer 0: gates[16 x 64 per wave] = x_t Wih0^T + h0 Whh0^T + b ----
    F32Acc acc[4];
#pragma unroll
    for (int G = 0; G < 4; ++G) {
      const int nt = 2 * G + wv;
      const v16h w0a = fetch_frag(sh_wcache, nt * 2 + 0, lane);
      const v16h w0b = fetch_frag(sh_wcache, nt * 2 + 1, lane);
#pragma unroll
      for (int r = 0; r < 8; ++r) acc[G].f[r] = bias0[G];
      acc[G].v = wmma_f16(xa0,   w0a,      acc[G].v);
      acc[G].v = wmma_f16(xa1,   w0b,      acc[G].v);
      acc[G].v = wmma_f16(h0f.v, whh0B[G], acc[G].v);
    }
#pragma unroll
    for (int r = 0; r < 8; ++r) {
      const float ig = sigm_pre(acc[0].f[r]);
      const float fg = sigm_pre(acc[1].f[r]);
      const float gg = tanh_fast(acc[2].f[r]);
      const float og = sigm_pre(acc[3].f[r]);
      c0[r] = fg * c0[r] + ig * gg;
      const float hv = og * tanh_fast(c0[r]);
      sh_h0[(half * 8 + r) * 32 + hcol] = (_Float16)hv;   // C-layout -> row-major LDS
    }
    __syncthreads();
    // row-major LDS -> A-frag layout (both waves read the full 16x32 tile)
    h0f.q[0] = *(const uint4*)&sh_h0[mrow * 32 + 8 * half];
    h0f.q[1] = *(const uint4*)&sh_h0[mrow * 32 + 16 + 8 * half];

    // ---- layer 1: gates = h0_t Wih1^T + h1 Whh1^T + b ----
#pragma unroll
    for (int G = 0; G < 4; ++G) {
      const int nt = 2 * G + wv;
      const v16h w1a = fetch_frag(sh_wcache, 16 + nt, lane);
      const v16h w1b = fetch_frag(sh_wcache, 24 + nt, lane);
#pragma unroll
      for (int r = 0; r < 8; ++r) acc[G].f[r] = bias1[G];
      acc[G].v = wmma_f16(h0f.v, w1a, acc[G].v);
      acc[G].v = wmma_f16(h1f.v, w1b, acc[G].v);
    }
#pragma unroll
    for (int r = 0; r < 8; ++r) {
      const float ig = sigm_pre(acc[0].f[r]);
      const float fg = sigm_pre(acc[1].f[r]);
      const float gg = tanh_fast(acc[2].f[r]);
      const float og = sigm_pre(acc[3].f[r]);
      c1[r] = fg * c1[r] + ig * gg;
      const float hv = og * tanh_fast(c1[r]);
      h1last[r] = hv;
      sh_h1[(half * 8 + r) * 32 + hcol] = (_Float16)hv;
    }
    __syncthreads();
    h1f.q[0] = *(const uint4*)&sh_h1[mrow * 32 + 8 * half];
    h1f.q[1] = *(const uint4*)&sh_h1[mrow * 32 + 16 + 8 * half];

    xa0 = nx0;
    xa1 = nx1;
  }

  // ---- final FC: out[b, :] = h1_T @ fc_w^T + fc_b  (OUT=2) ----
#pragma unroll
  for (int r = 0; r < 8; ++r) sh_f[(half * 8 + r) * 32 + hcol] = h1last[r];
  __syncthreads();
  if (threadIdx.x < 16) {
    const int m = threadIdx.x;
    float a0 = fc_b[0], a1 = fc_b[1];
#pragma unroll
    for (int h = 0; h < HID; ++h) {
      const float v = sh_f[m * 32 + h];
      a0 += v * fc_w[h];
      a1 += v * fc_w[HID + h];
    }
    out[(b0 + m) * 2 + 0] = a0;
    out[(b0 + m) * 2 + 1] = a1;
  }
}

} // namespace

extern "C" void kernel_launch(void* const* d_in, const int* in_sizes, int n_in,
                              void* d_out, int out_size, void* d_ws, size_t ws_size,
                              hipStream_t stream) {
  (void)n_in; (void)d_ws; (void)ws_size; (void)out_size;
  const float* x    = (const float*)d_in[0];
  const float* Wih0 = (const float*)d_in[1];
  const float* Whh0 = (const float*)d_in[2];
  const float* bih0 = (const float*)d_in[3];
  const float* bhh0 = (const float*)d_in[4];
  const float* Wih1 = (const float*)d_in[5];
  const float* Whh1 = (const float*)d_in[6];
  const float* bih1 = (const float*)d_in[7];
  const float* bhh1 = (const float*)d_in[8];
  const float* fc_w = (const float*)d_in[9];
  const float* fc_b = (const float*)d_in[10];
  float* out = (float*)d_out;

  const int B = in_sizes[0] / (T_STEPS * IN_DIM);  // 2048
  dim3 grid(B / 16);
  dim3 block(64);   // 2 waves: each owns half of the hidden columns
  hipLaunchKernelGGL(lstm2_fused, grid, block, 0, stream,
                     x, Wih0, Whh0, bih0, bhh0, Wih1, Whh1, bih1, bhh1, fc_w, fc_b, out);
}
